// UniDirectionalLSTM_25890062860961
// MI455X (gfx1250) — matrix operand order repaired and checked by reading, not compiled, not run on hardware
//
#include <hip/hip_runtime.h>
#include <cstdint>
#include <cstddef>

#define B_ 128
#define T_ 256
#define D_ 512
#define H_ 1024
#define C_ 7
#define G_ 4096   // 4*H

typedef __bf16 bf16;
typedef __attribute__((ext_vector_type(16))) __bf16 v16bf;
typedef __attribute__((ext_vector_type(8)))  __bf16 v8bf;
typedef __attribute__((ext_vector_type(8)))  float  v8f;

// ---------- WMMA helpers ----------

// A-operand load (16-bit A 16x32 packing): lane holds K = half*8+{0..7} and 16+half*8+{0..7}
// 'p' already includes row base + half*8; load p[0..7] and p[16..23].
__device__ __forceinline__ v16bf ld_a16(const bf16* p) {
    v8bf lo = *(const v8bf*)(p);
    v8bf hi = *(const v8bf*)(p + 16);
    return __builtin_shufflevector(lo, hi, 0,1,2,3,4,5,6,7,8,9,10,11,12,13,14,15);
}

// B-operand load (32x16 B packing): lane holds K = half*8+{0..7} and 16+half*8+{0..7}
// 'p' already includes row base + half*8; load p[0..7] and p[16..23].
__device__ __forceinline__ v16bf ld_b16(const bf16* p) {
    v8bf lo = *(const v8bf*)(p);
    v8bf hi = *(const v8bf*)(p + 16);
    return __builtin_shufflevector(lo, hi, 0,1,2,3,4,5,6,7,8,9,10,11,12,13,14,15);
}

__device__ __forceinline__ v8f wmma_bf16(v16bf a, v16bf b, v8f c) {
    // 8 args: (neg_a, A, neg_b, B, c_mod, C, reuse_a, reuse_b)
    return __builtin_amdgcn_wmma_f32_16x16x32_bf16(false, a, false, b, (short)0, c, false, false);
}

__device__ __forceinline__ float sigmoidf_(float x) { return 1.0f / (1.0f + expf(-x)); }

// Software-pipelined K-loop: operands for chunk k+1 are loaded into fresh
// registers before the 4 WMMAs of chunk k issue, keeping ~10 loads in flight.
__device__ __forceinline__ void gemm_k4(
    const bf16* __restrict__ arow,
    const bf16* __restrict__ b0, const bf16* __restrict__ b1,
    const bf16* __restrict__ b2, const bf16* __restrict__ b3,
    int K, v8f& accI, v8f& accF, v8f& accG, v8f& accO)
{
    v16bf a  = ld_a16(arow);
    v16bf w0 = ld_b16(b0), w1 = ld_b16(b1), w2 = ld_b16(b2), w3 = ld_b16(b3);
#pragma unroll 2
    for (int kc = 32; kc < K; kc += 32) {
        v16bf an = ld_a16(arow + kc);
        v16bf n0 = ld_b16(b0 + kc);
        v16bf n1 = ld_b16(b1 + kc);
        v16bf n2 = ld_b16(b2 + kc);
        v16bf n3 = ld_b16(b3 + kc);
        accI = wmma_bf16(a, w0, accI);
        accF = wmma_bf16(a, w1, accF);
        accG = wmma_bf16(a, w2, accG);
        accO = wmma_bf16(a, w3, accO);
        a = an; w0 = n0; w1 = n1; w2 = n2; w3 = n3;
    }
    accI = wmma_bf16(a, w0, accI);
    accF = wmma_bf16(a, w1, accF);
    accG = wmma_bf16(a, w2, accG);
    accO = wmma_bf16(a, w3, accO);
}

// ---------- small utility kernels ----------

__global__ void cvt_f32_bf16(const float* __restrict__ s, bf16* __restrict__ d, int n) {
    for (int i = blockIdx.x * blockDim.x + threadIdx.x; i < n; i += gridDim.x * blockDim.x)
        d[i] = (bf16)s[i];
}

__global__ void bias_sum(const float* __restrict__ a, const float* __restrict__ b,
                         float* __restrict__ d, int n) {
    int i = blockIdx.x * blockDim.x + threadIdx.x;
    if (i < n) d[i] = a[i] + b[i];
}

__global__ void zero_state(bf16* __restrict__ h, float* __restrict__ c, int n) {
    int i = blockIdx.x * blockDim.x + threadIdx.x;
    if (i < n) { h[i] = (bf16)0.0f; c[i] = 0.0f; }
}

// ---------- fused LSTM step ----------
// gates[b][g] = sum_k in[b,k]*w_ih[g,k] + sum_k h_prev[b,k]*w_hh[g,k] + bias[g]
// Each wave: one 16x16 (M x j) tile across ALL FOUR gates (4 accumulators),
// then the full cell update is done in-register.
// Grid: 64 blocks x 256 threads (8 waves). wave w = bid*8+wib; mt = w>>6 (0..7), jt = w&63.
template <int WRITE_SEQ>
__global__ __launch_bounds__(256) void lstm_step(
    const bf16* __restrict__ inA, long lda_in, int K_in,
    const bf16* __restrict__ w_ih,   // [4096 x K_in] row-major
    const bf16* __restrict__ w_hh,   // [4096 x 1024] row-major
    const float* __restrict__ bias,  // [4096]
    const bf16* __restrict__ h_prev, // [128 x 1024]
    bf16* __restrict__ h_next,       // [128 x 1024]
    float* __restrict__ c_state,     // [128 x 1024] (in-place)
    bf16* __restrict__ seq_out,      // already offset by t*H; stride = seq_stride
    long seq_stride)
{
    const int tid  = threadIdx.x;
    const int lane = tid & 31;
    const int wib  = tid >> 5;
    const int w    = blockIdx.x * 8 + wib;
    const int mt   = w >> 6;   // 0..7  (same for all waves in a block)
    const int jt   = w & 63;   // 0..63
    const int m_base = mt * 16;

    const int K_tot = K_in + H_;
    const int ldsw  = K_tot + 8;        // +8 bf16 (16B) row padding for bank spread
    extern __shared__ bf16 smem[];      // 16 rows x ldsw

    // ---- cooperative global -> LDS load of the 16-row A strip (input | h_prev) ----
    const int kt8 = K_tot >> 3;         // 8-elem chunks per row
    for (int idx = tid; idx < 16 * kt8; idx += 256) {
        int row = idx / kt8;
        int k   = (idx - row * kt8) << 3;
        v8bf v;
        if (k < K_in)
            v = *(const v8bf*)(inA + (size_t)(m_base + row) * lda_in + k);
        else
            v = *(const v8bf*)(h_prev + (size_t)(m_base + row) * H_ + (k - K_in));
        *(v8bf*)(&smem[row * ldsw + k]) = v;
    }
    __syncthreads();

    const int r    = lane & 15;
    const int half = lane >> 4;

    // A read base in LDS (A-packing offset half*8 baked in)
    const bf16* arow = smem + r * ldsw + half * 8;

    // B read bases: weight rows q*1024 + jt*16 + r act as columns (B-packing: +half*8)
    const int gcol = jt * 16 + r;
    const bf16* bi0 = w_ih + (size_t)(0 * H_ + gcol) * K_in + half * 8;
    const bf16* bi1 = w_ih + (size_t)(1 * H_ + gcol) * K_in + half * 8;
    const bf16* bi2 = w_ih + (size_t)(2 * H_ + gcol) * K_in + half * 8;
    const bf16* bi3 = w_ih + (size_t)(3 * H_ + gcol) * K_in + half * 8;
    const bf16* bh0 = w_hh + (size_t)(0 * H_ + gcol) * H_ + half * 8;
    const bf16* bh1 = w_hh + (size_t)(1 * H_ + gcol) * H_ + half * 8;
    const bf16* bh2 = w_hh + (size_t)(2 * H_ + gcol) * H_ + half * 8;
    const bf16* bh3 = w_hh + (size_t)(3 * H_ + gcol) * H_ + half * 8;

    v8f accI = {}, accF = {}, accG = {}, accO = {};

    // ---- input contribution (K_in), then recurrent contribution (H) ----
    gemm_k4(arow,          bi0, bi1, bi2, bi3, K_in, accI, accF, accG, accO);
    gemm_k4(arow + K_in,   bh0, bh1, bh2, bh3, H_,   accI, accF, accG, accO);

    // ---- fused LSTM cell update (in-register; each lane owns 8 (b,j) cells) ----
    const int j  = jt * 16 + r;            // hidden column (C/D layout: N = lane%16)
    const float bI = bias[0 * H_ + j];
    const float bF = bias[1 * H_ + j];
    const float bG = bias[2 * H_ + j];
    const float bO = bias[3 * H_ + j];

#pragma unroll
    for (int e = 0; e < 8; ++e) {
        const int b = m_base + e + 8 * half;   // C/D layout: M = e (+8 for lanes 16..31)
        const size_t idx = (size_t)b * H_ + j;
        float iv = accI[e] + bI;
        float fv = accF[e] + bF;
        float gv = accG[e] + bG;
        float ov = accO[e] + bO;
        float cp = c_state[idx];
        float cn = sigmoidf_(fv) * cp + sigmoidf_(iv) * tanhf(gv);
        float hn = sigmoidf_(ov) * tanhf(cn);
        c_state[idx] = cn;
        bf16 hb = (bf16)hn;
        h_next[idx] = hb;
        if (WRITE_SEQ) seq_out[(size_t)b * seq_stride + j] = hb;
    }
}

// ---------- final FC: out[b][c] = h[b,:] . fc_w[c,:] + fc_b[c] ----------
__global__ void fc_kernel(const bf16* __restrict__ h, const float* __restrict__ fcw,
                          const float* __restrict__ fcb, float* __restrict__ out) {
    int idx = blockIdx.x * blockDim.x + threadIdx.x;
    if (idx >= B_ * C_) return;
    int b = idx / C_, c = idx - b * C_;
    float s = fcb[c];
    const bf16* hp = h + (size_t)b * H_;
    const float* wp = fcw + (size_t)c * H_;
#pragma unroll 8
    for (int k = 0; k < H_; ++k) s += (float)hp[k] * wp[k];
    out[idx] = s;
}

// ---------- host ----------
extern "C" void kernel_launch(void* const* d_in, const int* in_sizes, int n_in,
                              void* d_out, int out_size, void* d_ws, size_t ws_size,
                              hipStream_t stream) {
    (void)in_sizes; (void)n_in; (void)out_size; (void)ws_size;
    const float* x    = (const float*)d_in[0];
    const float* wih0 = (const float*)d_in[1];
    const float* whh0 = (const float*)d_in[2];
    const float* bih0 = (const float*)d_in[3];
    const float* bhh0 = (const float*)d_in[4];
    const float* wih1 = (const float*)d_in[5];
    const float* whh1 = (const float*)d_in[6];
    const float* bih1 = (const float*)d_in[7];
    const float* bhh1 = (const float*)d_in[8];
    const float* fcw  = (const float*)d_in[9];
    const float* fcb  = (const float*)d_in[10];
    float* out = (float*)d_out;

    char* p = (char*)d_ws;
    auto carve = [&](size_t bytes) { char* r = p; p += (bytes + 255) & ~(size_t)255; return r; };
    bf16*  xb    = (bf16*) carve(sizeof(bf16)  * (size_t)B_ * T_ * D_);
    bf16*  wih0b = (bf16*) carve(sizeof(bf16)  * (size_t)G_ * D_);
    bf16*  whh0b = (bf16*) carve(sizeof(bf16)  * (size_t)G_ * H_);
    bf16*  wih1b = (bf16*) carve(sizeof(bf16)  * (size_t)G_ * H_);
    bf16*  whh1b = (bf16*) carve(sizeof(bf16)  * (size_t)G_ * H_);
    float* bias0 = (float*)carve(sizeof(float) * G_);
    float* bias1 = (float*)carve(sizeof(float) * G_);
    bf16*  seq1  = (bf16*) carve(sizeof(bf16)  * (size_t)B_ * T_ * H_);
    bf16*  hbuf0 = (bf16*) carve(sizeof(bf16)  * (size_t)B_ * H_);
    bf16*  hbuf1 = (bf16*) carve(sizeof(bf16)  * (size_t)B_ * H_);
    float* cst   = (float*)carve(sizeof(float) * (size_t)B_ * H_);

    // precision conversion (bf16 operands, fp32 accumulate in WMMA)
    cvt_f32_bf16<<<2048, 256, 0, stream>>>(x,    xb,    B_ * T_ * D_);
    cvt_f32_bf16<<<1024, 256, 0, stream>>>(wih0, wih0b, G_ * D_);
    cvt_f32_bf16<<<1024, 256, 0, stream>>>(whh0, whh0b, G_ * H_);
    cvt_f32_bf16<<<1024, 256, 0, stream>>>(wih1, wih1b, G_ * H_);
    cvt_f32_bf16<<<1024, 256, 0, stream>>>(whh1, whh1b, G_ * H_);
    bias_sum<<<16, 256, 0, stream>>>(bih0, bhh0, bias0, G_);
    bias_sum<<<16, 256, 0, stream>>>(bih1, bhh1, bias1, G_);

    bf16* hb[2] = { hbuf0, hbuf1 };

    // ---- layer 0 ----
    zero_state<<<512, 256, 0, stream>>>(hbuf0, cst, B_ * H_);
    size_t shm0 = sizeof(bf16) * 16 * (size_t)(D_ + H_ + 8);
    for (int t = 0; t < T_; ++t) {
        lstm_step<1><<<64, 256, shm0, stream>>>(
            xb + (size_t)t * D_, (long)T_ * D_, D_,
            wih0b, whh0b, bias0,
            hb[t & 1], hb[(t + 1) & 1], cst,
            seq1 + (size_t)t * H_, (long)T_ * H_);
    }

    // ---- layer 1 ----
    zero_state<<<512, 256, 0, stream>>>(hbuf0, cst, B_ * H_);
    size_t shm1 = sizeof(bf16) * 16 * (size_t)(H_ + H_ + 8);
    for (int t = 0; t < T_; ++t) {
        lstm_step<0><<<64, 256, shm1, stream>>>(
            seq1 + (size_t)t * H_, (long)T_ * H_, H_,
            wih1b, whh1b, bias1,
            hb[t & 1], hb[(t + 1) & 1], cst,
            nullptr, 0);
    }

    // final hidden of layer 1 lives in hb[(T-1+1)&1] == hb[0]
    fc_kernel<<<(B_ * C_ + 255) / 256, 256, 0, stream>>>(hb[0], fcw, fcb, out);
}
